// MoE_27925877358890
// MI455X (gfx1250) — matrix operand order repaired and checked by reading, not compiled, not run on hardware
//
#include <hip/hip_runtime.h>

#define BDIM 4
#define SDIM 2048
#define HDIM 1024
#define EDIM 8
#define KSEL 2
#define NTOK (BDIM * SDIM)                 // 8192 tokens
#define CAP  NTOK                          // worst-case tokens per expert
#define MTILE 32
#define MAXTILES (NTOK * KSEL / MTILE + EDIM) // 520 M-tiles upper bound

typedef unsigned int u32x4 __attribute__((ext_vector_type(4)));
typedef unsigned int u32x2 __attribute__((ext_vector_type(2)));
typedef __bf16       v16bf __attribute__((ext_vector_type(16)));
typedef float        v8f   __attribute__((ext_vector_type(8)));

union FragBF { u32x4 q[2]; v16bf v; };

__device__ __forceinline__ unsigned short f2bf(float f) {
    unsigned u = __float_as_uint(f);
    u += 0x7fffu + ((u >> 16) & 1u);       // round-to-nearest-even
    return (unsigned short)(u >> 16);
}

// ---------------- zero kernels ----------------
__global__ void zero_meta(int* counts) {
    if (threadIdx.x < EDIM) counts[threadIdx.x] = 0;
}

__global__ void zero_out(float* __restrict__ out) {
    size_t i = ((size_t)blockIdx.x * blockDim.x + threadIdx.x) * 4;
    u32x4 z = {0u, 0u, 0u, 0u};
    *(u32x4*)(out + i) = z;
}

// ---------------- x: f32 -> bf16 ----------------
__global__ void cvt_x(const float* __restrict__ x, unsigned short* __restrict__ xb) {
    size_t i = ((size_t)blockIdx.x * blockDim.x + threadIdx.x) * 4;
    float a0 = x[i], a1 = x[i + 1], a2 = x[i + 2], a3 = x[i + 3];
    u32x2 p;
    p.x = (unsigned)f2bf(a0) | ((unsigned)f2bf(a1) << 16);
    p.y = (unsigned)f2bf(a2) | ((unsigned)f2bf(a3) << 16);
    *(u32x2*)(xb + i) = p;
}

// ---------------- W[e][h][o] f32 -> WbT[e][o][h] bf16 (LDS tile transpose) ----------------
__global__ void cvt_wT(const float* __restrict__ W, unsigned short* __restrict__ WbT) {
    __shared__ float tile[32][33];
    int ex = blockIdx.z;
    int hb = blockIdx.x * 32, ob = blockIdx.y * 32;
    int tx = threadIdx.x, ty = threadIdx.y;            // 32x8
    const float* src = W + ((size_t)ex * HDIM + hb) * HDIM + ob;
    for (int r = ty; r < 32; r += 8) tile[r][tx] = src[(size_t)r * HDIM + tx];
    __syncthreads();
    unsigned short* dst = WbT + ((size_t)ex * HDIM + ob) * HDIM + hb;
    for (int r = ty; r < 32; r += 8) dst[(size_t)r * HDIM + tx] = f2bf(tile[tx][r]);
}

// ---------------- gating: softmax + top-2 + routing lists (1 wave / token) ----------------
__global__ void moe_gate(const float* __restrict__ x, const float* __restrict__ Wg,
                         const float* __restrict__ bg, int* __restrict__ counts,
                         int* __restrict__ tokList, float* __restrict__ gateList) {
    int lane  = threadIdx.x & 31;
    int token = blockIdx.x * (blockDim.x >> 5) + (threadIdx.x >> 5);
    const float* xr = x + (size_t)token * HDIM;
    float acc[EDIM];
#pragma unroll
    for (int e = 0; e < EDIM; ++e) acc[e] = 0.f;
    for (int h = lane; h < HDIM; h += 32) {
        float xv = xr[h];
        const float* wr = Wg + (size_t)h * EDIM;
#pragma unroll
        for (int e = 0; e < EDIM; ++e) acc[e] = fmaf(xv, wr[e], acc[e]);
    }
#pragma unroll
    for (int off = 16; off >= 1; off >>= 1)
#pragma unroll
        for (int e = 0; e < EDIM; ++e) acc[e] += __shfl_xor(acc[e], off, 32);

    float lg[EDIM];
    float mx = -3.4e38f;
#pragma unroll
    for (int e = 0; e < EDIM; ++e) { lg[e] = acc[e] + bg[e]; mx = fmaxf(mx, lg[e]); }
    float s = 0.f;
#pragma unroll
    for (int e = 0; e < EDIM; ++e) { lg[e] = __expf(lg[e] - mx); s += lg[e]; }
    float inv = 1.f / s;
    int i1 = 0; float v1 = lg[0];
#pragma unroll
    for (int e = 1; e < EDIM; ++e) if (lg[e] > v1) { v1 = lg[e]; i1 = e; }
    int i2 = -1; float v2 = -1.f;
#pragma unroll
    for (int e = 0; e < EDIM; ++e) if (e != i1 && lg[e] > v2) { v2 = lg[e]; i2 = e; }
    if (lane == 0) {
        int s1 = atomicAdd(&counts[i1], 1);
        tokList[i1 * CAP + s1] = token; gateList[i1 * CAP + s1] = v1 * inv;
        int s2 = atomicAdd(&counts[i2], 1);
        tokList[i2 * CAP + s2] = token; gateList[i2 * CAP + s2] = v2 * inv;
    }
}

// ---------------- tile prefix over experts ----------------
__global__ void tile_prefix(const int* __restrict__ counts, int* __restrict__ tileOff) {
    tileOff[0] = 0;
    for (int e = 0; e < EDIM; ++e)
        tileOff[e + 1] = tileOff[e] + (counts[e] + MTILE - 1) / MTILE;
}

// ---------------- routed grouped GEMM: out[tok] += g * (x @ W[e] + b[e]) ----------------
// Block = 4 waves, tile = 32(M) x 128(N); each wave: 2 A-frags x 2 B-frags -> 4 WMMAs/K-step.
__global__ void __launch_bounds__(128) moe_gemm(
        const unsigned short* __restrict__ xb, const unsigned short* __restrict__ WbT,
        const float* __restrict__ bias, float* __restrict__ out,
        const int* __restrict__ counts, const int* __restrict__ tileOff,
        const int* __restrict__ tokList, const float* __restrict__ gateList) {
    int bx = blockIdx.x;
    if (bx >= tileOff[EDIM]) return;               // uniform early-exit
    int ex = 0, lt = 0;
#pragma unroll
    for (int i = 0; i < EDIM; ++i)
        if (bx >= tileOff[i] && bx < tileOff[i + 1]) { ex = i; lt = bx - tileOff[i]; }
    int cnt = counts[ex];

    __shared__ int   sTok[MTILE];
    __shared__ float sGate[MTILE];
    if (threadIdx.x < MTILE) {
        int pos = lt * MTILE + threadIdx.x;
        if (pos < cnt) { sTok[threadIdx.x] = tokList[ex * CAP + pos];
                         sGate[threadIdx.x] = gateList[ex * CAP + pos]; }
        else           { sTok[threadIdx.x] = 0; sGate[threadIdx.x] = 0.f; }
    }
    __syncthreads();

    int lane = threadIdx.x & 31;
    int wave = threadIdx.x >> 5;
    int hf   = lane >> 4;                          // half-wave select
    int nl   = lane & 15;

    // A: lane holds row nl (frag0) and row 16+nl (frag1)
    const unsigned short* aRow0 = xb + (size_t)sTok[nl] * HDIM;
    const unsigned short* aRow1 = xb + (size_t)sTok[16 + nl] * HDIM;
    int nBase = blockIdx.y * 128 + wave * 32;
    const unsigned short* b0Col = WbT + ((size_t)ex * HDIM + (nBase + nl)) * HDIM;
    const unsigned short* b1Col = b0Col + (size_t)16 * HDIM;

    v8f c00 = {0.f,0.f,0.f,0.f,0.f,0.f,0.f,0.f};
    v8f c01 = {0.f,0.f,0.f,0.f,0.f,0.f,0.f,0.f};
    v8f c10 = {0.f,0.f,0.f,0.f,0.f,0.f,0.f,0.f};
    v8f c11 = {0.f,0.f,0.f,0.f,0.f,0.f,0.f,0.f};

    for (int k0 = 0; k0 < HDIM; k0 += 32) {
        FragBF a0, a1, b0, b1;
        // A 16x32 bf16 layout: half 0 -> K[0..7]+K[16..23], half 1 -> K[8..15]+K[24..31]
        int ao = k0 + hf * 8;
        a0.q[0] = *(const u32x4*)(aRow0 + ao);
        a0.q[1] = *(const u32x4*)(aRow0 + ao + 16);
        a1.q[0] = *(const u32x4*)(aRow1 + ao);
        a1.q[1] = *(const u32x4*)(aRow1 + ao + 16);
        // B 32x16 bf16 layout: lane holds column, half 0 -> K[0..7]+K[16..23], half 1 -> K[8..15]+K[24..31]
        int bo = k0 + hf * 8;
        b0.q[0] = *(const u32x4*)(b0Col + bo);
        b0.q[1] = *(const u32x4*)(b0Col + bo + 16);
        b1.q[0] = *(const u32x4*)(b1Col + bo);
        b1.q[1] = *(const u32x4*)(b1Col + bo + 16);
        c00 = __builtin_amdgcn_wmma_f32_16x16x32_bf16(false, a0.v, false, b0.v,
                                                      (short)0, c00, false, false);
        c01 = __builtin_amdgcn_wmma_f32_16x16x32_bf16(false, a0.v, false, b1.v,
                                                      (short)0, c01, false, false);
        c10 = __builtin_amdgcn_wmma_f32_16x16x32_bf16(false, a1.v, false, b0.v,
                                                      (short)0, c10, false, false);
        c11 = __builtin_amdgcn_wmma_f32_16x16x32_bf16(false, a1.v, false, b1.v,
                                                      (short)0, c11, false, false);
    }

    int n0 = nBase + nl, n1 = n0 + 16;
    float bias0 = bias[ex * HDIM + n0];
    float bias1 = bias[ex * HDIM + n1];
#pragma unroll
    for (int i = 0; i < 8; ++i) {
        int m = hf * 8 + i;                        // C layout: half-wave selects M block
        float g0 = sGate[m];
        if (g0 != 0.f) {
            size_t rb = (size_t)sTok[m] * HDIM;
            unsafeAtomicAdd(out + rb + n0, g0 * (c00[i] + bias0));
            unsafeAtomicAdd(out + rb + n1, g0 * (c01[i] + bias1));
        }
        float g1 = sGate[16 + m];
        if (g1 != 0.f) {
            size_t rb = (size_t)sTok[16 + m] * HDIM;
            unsafeAtomicAdd(out + rb + n0, g1 * (c10[i] + bias0));
            unsafeAtomicAdd(out + rb + n1, g1 * (c11[i] + bias1));
        }
    }
}

// ---------------- launcher ----------------
extern "C" void kernel_launch(void* const* d_in, const int* in_sizes, int n_in,
                              void* d_out, int out_size, void* d_ws, size_t ws_size,
                              hipStream_t stream) {
    const float* x  = (const float*)d_in[0];   // [B,S,H]
    const float* W  = (const float*)d_in[1];   // [E,H,H]
    const float* b  = (const float*)d_in[2];   // [E,H]
    const float* Wg = (const float*)d_in[3];   // [H,E]
    const float* bg = (const float*)d_in[4];   // [E]
    float* out = (float*)d_out;

    char* ws = (char*)d_ws;
    int*   counts   = (int*)ws;                                    // 8 ints
    int*   tileOff  = (int*)(ws + 64);                             // 9 ints
    int*   tokList  = (int*)(ws + 256);                            // E*CAP ints
    float* gateList = (float*)(ws + 256 + (size_t)EDIM * CAP * 4); // E*CAP floats
    unsigned short* xb  = (unsigned short*)(ws + (1u << 20));
    unsigned short* WbT = (unsigned short*)(ws + (1u << 20) + (size_t)NTOK * HDIM * 2);

    zero_meta<<<1, 32, 0, stream>>>(counts);
    zero_out <<<NTOK * HDIM / (256 * 4), 256, 0, stream>>>(out);
    cvt_x    <<<NTOK * HDIM / (256 * 4), 256, 0, stream>>>(x, xb);
    cvt_wT   <<<dim3(HDIM / 32, HDIM / 32, EDIM), dim3(32, 8), 0, stream>>>(W, WbT);
    moe_gate <<<NTOK / 4, 128, 0, stream>>>(x, Wg, bg, counts, tokList, gateList);
    tile_prefix<<<1, 1, 0, stream>>>(counts, tileOff);
    moe_gemm <<<dim3(MAXTILES, HDIM / 128), 128, 0, stream>>>(
        xb, WbT, b, out, counts, tileOff, tokList, gateList);
}
